// SAB_65712999628857
// MI455X (gfx1250) — matrix operand order repaired and checked by reading, not compiled, not run on hardware
//
#include <hip/hip_runtime.h>
#include <hip/hip_bf16.h>

typedef __bf16 bf16;
typedef __attribute__((ext_vector_type(16))) __bf16 bf16x16;
typedef __attribute__((ext_vector_type(8)))  float  f32x8;

#define B_SZ    8
#define N_TOK   1024
#define D_MODEL 1024
#define H_HEADS 16
#define T_DIM   64
#define DOUT    1024
#define ROWS    (B_SZ * N_TOK)     // 8192
#define QKV_PITCH (3 * DOUT)       // 3072

union FragU {
  bf16x16  v;
  unsigned u[8];
  uint4    q[2];
};

// ---------------------------------------------------------------------------
// VALU (DPP) row reductions across the 16-lane half-row: rotate-and-combine
// with row_ror:1/2/4/8 leaves the full 16-lane result in every lane.
// ---------------------------------------------------------------------------
template <int CTRL>
__device__ __forceinline__ float dpp_rotf(float x) {
  int xi = __float_as_int(x);
  int r = __builtin_amdgcn_update_dpp(xi, xi, CTRL, 0xF, 0xF, false);
  return __int_as_float(r);
}
__device__ __forceinline__ float row_max16(float v) {
  v = fmaxf(v, dpp_rotf<0x121>(v));   // row_ror:1
  v = fmaxf(v, dpp_rotf<0x122>(v));   // row_ror:2
  v = fmaxf(v, dpp_rotf<0x124>(v));   // row_ror:4
  v = fmaxf(v, dpp_rotf<0x128>(v));   // row_ror:8
  return v;
}
__device__ __forceinline__ float row_sum16(float v) {
  v += dpp_rotf<0x121>(v);
  v += dpp_rotf<0x122>(v);
  v += dpp_rotf<0x124>(v);
  v += dpp_rotf<0x128>(v);
  return v;
}

// ---------------------------------------------------------------------------
// Repack / convert kernels
// ---------------------------------------------------------------------------
__global__ void cvt_f32_to_bf16(const float* __restrict__ s, bf16* __restrict__ d, int n) {
  int i = blockIdx.x * blockDim.x + threadIdx.x;
  if (i < n) d[i] = (bf16)s[i];
}

// Wq/Wk/Wv [H, D, T] fp32  ->  Wqkv_bf [D, 3*DOUT] bf16, col = sel*1024 + h*64 + t
__global__ void repack_wqkv(const float* __restrict__ Wq, const float* __restrict__ Wk,
                            const float* __restrict__ Wv, bf16* __restrict__ dst) {
  int i = blockIdx.x * blockDim.x + threadIdx.x;
  const int per = D_MODEL * DOUT;
  if (i >= 3 * per) return;
  int sel = i / per;
  int rem = i - sel * per;                  // h*D*T + d*T + t
  int hh = rem / (D_MODEL * T_DIM);
  int r2 = rem - hh * (D_MODEL * T_DIM);
  int d  = r2 / T_DIM;
  int t  = r2 - d * T_DIM;
  const float* W = (sel == 0) ? Wq : (sel == 1 ? Wk : Wv);
  dst[(size_t)d * QKV_PITCH + sel * DOUT + hh * T_DIM + t] = (bf16)W[rem];
}

__global__ void pack_bias(const float* __restrict__ bq, const float* __restrict__ bk,
                          const float* __restrict__ bv, float* __restrict__ dst) {
  int i = blockIdx.x * blockDim.x + threadIdx.x;
  if (i >= 3 * DOUT) return;
  int sel = i / DOUT, c = i - sel * DOUT;
  dst[i] = (sel == 0) ? bq[c] : (sel == 1 ? bk[c] : bv[c]);
}

// ---------------------------------------------------------------------------
// WMMA GEMM: C[M,N] = A[M,K] (bf16, rm) @ B[K,N] (bf16, rm) + bias
// EP: 0 = bias -> bf16 ; 1 = bias+relu -> bf16 ; 2 = bias -> f32
// 128x128x32 tiles, 8 waves x (64x32), double-buffered LDS with
// load(k+1) -> compute(k) -> store(k+1) pipelining to hide VMEM latency.
// A-fragments are loaded per-mt to keep register pressure low (no spills).
// ---------------------------------------------------------------------------
template <int EP>
__global__ __launch_bounds__(256)
void wmma_gemm(const bf16* __restrict__ A, const bf16* __restrict__ Bm,
               const float* __restrict__ bias, void* __restrict__ Cout,
               int M, int Nn, int K) {
  __shared__ bf16 ldsA[2][128 * 40];        // row-major, pitch 40 bf16
  __shared__ bf16 ldsBt[2][128 * 40];       // B transposed: [n][k], pitch 40

  const int tid  = threadIdx.x;
  const int lane = tid & 31;
  const int wave = tid >> 5;
  const int lm = lane & 15;
  const int g  = lane >> 4;
  const int wm = wave >> 2;                 // 0..1
  const int wn = wave & 3;                  // 0..3
  const int rowBase = blockIdx.y * 128;
  const int colBase = blockIdx.x * 128;
  const int NK = K >> 5;

  uint4 ra[2], rb[2];

  auto loadTiles = [&](int kt) {
    const int kk = kt * 32;
#pragma unroll
    for (int j = 0; j < 2; ++j) {
      int c = tid + 256 * j;
      int arow = c >> 2, acol = (c & 3) * 8;
      ra[j] = *(const uint4*)(A + (size_t)(rowBase + arow) * K + kk + acol);
      int kr = c >> 4, bcol = (c & 15) * 8;
      rb[j] = *(const uint4*)(Bm + (size_t)(kk + kr) * Nn + colBase + bcol);
    }
  };
  auto storeTiles = [&](int buf) {
#pragma unroll
    for (int j = 0; j < 2; ++j) {
      int c = tid + 256 * j;
      int arow = c >> 2, acol = (c & 3) * 8;
      *(uint4*)&ldsA[buf][arow * 40 + acol] = ra[j];
      int kr = c >> 4, bcol = (c & 15) * 8;
      union { uint4 q; bf16 e[8]; } t;
      t.q = rb[j];
#pragma unroll
      for (int e = 0; e < 8; ++e) ldsBt[buf][(bcol + e) * 40 + kr] = t.e[e];
    }
  };

  const f32x8 zero = {0.f, 0.f, 0.f, 0.f, 0.f, 0.f, 0.f, 0.f};
  f32x8 acc[4][2];
#pragma unroll
  for (int mt = 0; mt < 4; ++mt)
#pragma unroll
    for (int nt = 0; nt < 2; ++nt) acc[mt][nt] = zero;

  loadTiles(0);
  storeTiles(0);
  __syncthreads();

  for (int kt = 0; kt < NK; ++kt) {
    const int cur = kt & 1;
    if (kt + 1 < NK) loadTiles(kt + 1);     // global loads in flight over wmma

    FragU b0, b1;
    {
      const bf16* base = &ldsBt[cur][(wn * 32 + lm) * 40 + g * 8];
      b0.q[0] = *(const uint4*)(base);
      b0.q[1] = *(const uint4*)(base + 16);
      base += 16 * 40;
      b1.q[0] = *(const uint4*)(base);
      b1.q[1] = *(const uint4*)(base + 16);
    }
#pragma unroll
    for (int mt = 0; mt < 4; ++mt) {
      FragU a;
      const bf16* base = &ldsA[cur][(wm * 64 + mt * 16 + lm) * 40];
#pragma unroll
      for (int p = 0; p < 8; ++p) {
        int k0 = ((p < 4) ? 0 : 16) + g * 8 + 2 * (p & 3);
        a.u[p] = *(const unsigned*)(base + k0);
      }
      acc[mt][0] = __builtin_amdgcn_wmma_f32_16x16x32_bf16(
          false, a.v, false, b0.v, (short)0, acc[mt][0], false, false);
      acc[mt][1] = __builtin_amdgcn_wmma_f32_16x16x32_bf16(
          false, a.v, false, b1.v, (short)0, acc[mt][1], false, false);
    }

    if (kt + 1 < NK) storeTiles(cur ^ 1);   // waits loadcnt only here
    __syncthreads();
  }

#pragma unroll
  for (int mt = 0; mt < 4; ++mt) {
#pragma unroll
    for (int nt = 0; nt < 2; ++nt) {
      const int col = colBase + wn * 32 + nt * 16 + lm;
      const float bv = bias[col];
#pragma unroll
      for (int r = 0; r < 8; ++r) {
        const int row = rowBase + wm * 64 + mt * 16 + r + 8 * g;
        float v = acc[mt][nt][r] + bv;
        if (EP == 1) v = fmaxf(v, 0.f);
        if (EP == 2)
          ((float*)Cout)[(size_t)row * Nn + col] = v;
        else
          ((bf16*)Cout)[(size_t)row * Nn + col] = (bf16)v;
      }
    }
  }
}

// ---------------------------------------------------------------------------
// Flash attention per (b, h, 64-row q block); 4 waves x 16 q rows each.
// QKV layout: [b*N + n][3072] bf16, Q at +0, K at +1024, V at +2048 (+h*64)
// Out: [b*N + n][1024] fp32, channel = t*H + h (reference transpose order)
// LDS ping-pong (one barrier/iter); load+store fused so prefetch registers
// have minimal live range (no scratch spills); occupancy hides VMEM latency.
// ---------------------------------------------------------------------------
__global__ __launch_bounds__(128)
void attn_kernel(const bf16* __restrict__ QKV, float* __restrict__ Out) {
  const int qb = blockIdx.x;
  const int h  = blockIdx.y;
  const int b  = blockIdx.z;
  const int tid  = threadIdx.x;
  const int lane = tid & 31;
  const int wave = tid >> 5;
  const int lm = lane & 15;
  const int g  = lane >> 4;

  __shared__ bf16 ldsK[2][32 * 72];         // K chunk [kv][d], pitch 72
  __shared__ bf16 ldsVt[2][64 * 40];        // V chunk transposed [t][kv], pitch 40
  __shared__ bf16 ldsP[4][16 * 34];         // per-wave P tile [q][kv], pitch 34

  const bf16* Qbase = QKV + (size_t)(b * N_TOK) * QKV_PITCH + h * T_DIM;
  const bf16* Kbase = Qbase + DOUT;
  const bf16* Vbase = Qbase + 2 * DOUT;

  auto stageKV = [&](int ck, int buf) {     // global -> LDS, short reg lifetime
    const int kv0 = ck * 32;
#pragma unroll
    for (int j = 0; j < 2; ++j) {
      int c = tid + 128 * j;                // 0..255 chunks of 8 bf16
      int row = c >> 3;                     // 0..31 kv row
      int col8 = (c & 7) * 8;
      uint4 kq = *(const uint4*)(Kbase + (size_t)(kv0 + row) * QKV_PITCH + col8);
      union { uint4 q; bf16 e[8]; } t;
      t.q = *(const uint4*)(Vbase + (size_t)(kv0 + row) * QKV_PITCH + col8);
      *(uint4*)&ldsK[buf][row * 72 + col8] = kq;
#pragma unroll
      for (int e = 0; e < 8; ++e) ldsVt[buf][(col8 + e) * 40 + row] = t.e[e];
    }
  };

  // Q fragments (K-dim = d, two 32-chunks), held in registers for whole pass
  const int qrow = qb * 64 + wave * 16 + lm;
  FragU qa[2];
#pragma unroll
  for (int c = 0; c < 2; ++c)
#pragma unroll
    for (int p = 0; p < 8; ++p) {
      int k0 = c * 32 + ((p < 4) ? 0 : 16) + g * 8 + 2 * (p & 3);
      qa[c].u[p] = *(const unsigned*)(Qbase + (size_t)qrow * QKV_PITCH + k0);
    }

  const f32x8 zero = {0.f, 0.f, 0.f, 0.f, 0.f, 0.f, 0.f, 0.f};
  f32x8 o[4];
#pragma unroll
  for (int j = 0; j < 4; ++j) o[j] = zero;
  float m[8], l[8];
#pragma unroll
  for (int r = 0; r < 8; ++r) { m[r] = -3.0e38f; l[r] = 0.f; }

  const int NC = N_TOK / 32;
  stageKV(0, 0);
  __syncthreads();

  for (int ck = 0; ck < NC; ++ck) {
    const int cur = ck & 1;
    if (ck + 1 < NC) stageKV(ck + 1, cur ^ 1);

    // S tile: 16q x 32kv, contraction over d=64 (two wmma per 16-col tile)
    f32x8 s01[2];
    s01[0] = zero; s01[1] = zero;
#pragma unroll
    for (int nt = 0; nt < 2; ++nt) {
#pragma unroll
      for (int c = 0; c < 2; ++c) {
        FragU kb;
        const bf16* base = &ldsK[cur][(nt * 16 + lm) * 72 + c * 32 + g * 8];
        kb.q[0] = *(const uint4*)(base);
        kb.q[1] = *(const uint4*)(base + 16);
        s01[nt] = __builtin_amdgcn_wmma_f32_16x16x32_bf16(
            false, qa[c].v, false, kb.v, (short)0, s01[nt], false, false);
      }
    }

    // Online softmax update (scale = 1/sqrt(H*TV) = 1/32), all-VALU reductions
    const float scale = 0.03125f;
#pragma unroll
    for (int r = 0; r < 8; ++r) {
      float s0 = s01[0][r] * scale;
      float s1 = s01[1][r] * scale;
      float mx = row_max16(fmaxf(s0, s1));
      float nm = fmaxf(m[r], mx);
      float alpha = __expf(m[r] - nm);
      float p0 = __expf(s0 - nm);
      float p1 = __expf(s1 - nm);
      float rs = row_sum16(p0 + p1);
      l[r] = l[r] * alpha + rs;
      m[r] = nm;
#pragma unroll
      for (int j = 0; j < 4; ++j) o[j][r] *= alpha;
      const int prow = r + 8 * g;
      ldsP[wave][prow * 34 + lm]      = (bf16)p0;
      ldsP[wave][prow * 34 + 16 + lm] = (bf16)p1;
    }

    // P (16x32) as A-fragment; O += P @ V over four 16-col tv tiles
    FragU pa;
#pragma unroll
    for (int p = 0; p < 8; ++p) {
      int k0 = ((p < 4) ? 0 : 16) + g * 8 + 2 * (p & 3);
      pa.u[p] = *(const unsigned*)(&ldsP[wave][lm * 34 + k0]);
    }
#pragma unroll
    for (int j = 0; j < 4; ++j) {
      FragU vb;
      const bf16* base = &ldsVt[cur][(j * 16 + lm) * 40 + g * 8];
      vb.q[0] = *(const uint4*)(base);
      vb.q[1] = *(const uint4*)(base + 16);
      o[j] = __builtin_amdgcn_wmma_f32_16x16x32_bf16(
          false, pa.v, false, vb.v, (short)0, o[j], false, false);
    }

    __syncthreads();
  }

#pragma unroll
  for (int j = 0; j < 4; ++j)
#pragma unroll
    for (int r = 0; r < 8; ++r) {
      const int row = qb * 64 + wave * 16 + r + 8 * g;
      const int t = j * 16 + lm;
      Out[((size_t)(b * N_TOK) + row) * DOUT + t * H_HEADS + h] = o[j][r] / l[r];
    }
}

// ---------------------------------------------------------------------------
// Residual + LayerNorm: y = LN(xa + xb) * gamma + beta; writes f32 (+opt bf16)
// One row (1024 cols) per 256-thread block.
// ---------------------------------------------------------------------------
__global__ __launch_bounds__(256)
void ln_kernel(const float* __restrict__ xa, const float* __restrict__ xb,
               const float* __restrict__ gamma, const float* __restrict__ beta,
               float* __restrict__ outf, bf16* __restrict__ outb) {
  const int row = blockIdx.x;
  const int tid = threadIdx.x;
  const int lane = tid & 31, wave = tid >> 5;
  const float* pa = xa + (size_t)row * DOUT;
  const float* pb = xb + (size_t)row * DOUT;

  __shared__ float red[8];
  __shared__ float s_mu, s_inv;

  float v[4];
  float s = 0.f;
#pragma unroll
  for (int i = 0; i < 4; ++i) {
    int c = tid + i * 256;
    v[i] = pa[c] + pb[c];
    s += v[i];
  }
  s = row_sum16(s);
  s += __shfl_xor(s, 16, 32);               // cross-half combine
  if (lane == 0) red[wave] = s;
  __syncthreads();
  if (tid == 0) {
    float t = 0.f;
    for (int w = 0; w < 8; ++w) t += red[w];
    s_mu = t * (1.f / DOUT);
  }
  __syncthreads();
  const float mu = s_mu;
  float q = 0.f;
#pragma unroll
  for (int i = 0; i < 4; ++i) { float d = v[i] - mu; q += d * d; }
  q = row_sum16(q);
  q += __shfl_xor(q, 16, 32);
  if (lane == 0) red[wave] = q;
  __syncthreads();
  if (tid == 0) {
    float t = 0.f;
    for (int w = 0; w < 8; ++w) t += red[w];
    s_inv = rsqrtf(t * (1.f / DOUT) + 1e-5f);
  }
  __syncthreads();
  const float inv = s_inv;
#pragma unroll
  for (int i = 0; i < 4; ++i) {
    int c = tid + i * 256;
    float y = (v[i] - mu) * inv * gamma[c] + beta[c];
    outf[(size_t)row * DOUT + c] = y;
    if (outb) outb[(size_t)row * DOUT + c] = (bf16)y;
  }
}

// ---------------------------------------------------------------------------
extern "C" void kernel_launch(void* const* d_in, const int* in_sizes, int n_in,
                              void* d_out, int out_size, void* d_ws, size_t ws_size,
                              hipStream_t stream) {
  (void)in_sizes; (void)n_in; (void)out_size; (void)ws_size;
  const float* tokens = (const float*)d_in[0];
  const float* Wq  = (const float*)d_in[1];
  const float* bq  = (const float*)d_in[2];
  const float* Wk  = (const float*)d_in[3];
  const float* bk  = (const float*)d_in[4];
  const float* Wv  = (const float*)d_in[5];
  const float* bv  = (const float*)d_in[6];
  const float* W1  = (const float*)d_in[7];
  const float* b1  = (const float*)d_in[8];
  const float* W2  = (const float*)d_in[9];
  const float* b2  = (const float*)d_in[10];
  const float* g1  = (const float*)d_in[11];
  const float* be1 = (const float*)d_in[12];
  const float* g2  = (const float*)d_in[13];
  const float* be2 = (const float*)d_in[14];

  char* ws = (char*)d_ws;
  size_t off = 0;
  auto take = [&](size_t bytes) -> char* {
    char* p = ws + off;
    off += (bytes + 255) & ~(size_t)255;
    return p;
  };

  bf16*  Xb    = (bf16*)take((size_t)ROWS * D_MODEL * 2);     // tokens bf16
  bf16*  Wqkvb = (bf16*)take((size_t)D_MODEL * QKV_PITCH * 2);
  float* bqkv  = (float*)take((size_t)QKV_PITCH * 4);
  bf16*  W1b   = (bf16*)take((size_t)DOUT * DOUT * 2);
  bf16*  W2b   = (bf16*)take((size_t)DOUT * DOUT * 2);
  bf16*  QKV   = (bf16*)take((size_t)ROWS * QKV_PITCH * 2);
  float* AO    = (float*)take((size_t)ROWS * DOUT * 4);
  bf16*  H1B   = (bf16*)take((size_t)ROWS * DOUT * 2);

  // Lifetime-safe aliases (stream-ordered):
  bf16*  R1B = Xb;           // Xb last read by QKV GEMM; reused for relu output
  float* H1F = (float*)QKV;  // QKV last read by attention; reused for LN1 f32 out
  float* FF  = AO;           // AO last read by LN1; reused for FFN2 output

  {
    int n = ROWS * D_MODEL;
    cvt_f32_to_bf16<<<(n + 255) / 256, 256, 0, stream>>>(tokens, Xb, n);
  }
  {
    int n = DOUT * DOUT;
    cvt_f32_to_bf16<<<(n + 255) / 256, 256, 0, stream>>>(W1, W1b, n);
    cvt_f32_to_bf16<<<(n + 255) / 256, 256, 0, stream>>>(W2, W2b, n);
  }
  repack_wqkv<<<(3 * D_MODEL * DOUT + 255) / 256, 256, 0, stream>>>(Wq, Wk, Wv, Wqkvb);
  pack_bias<<<(3 * DOUT + 255) / 256, 256, 0, stream>>>(bq, bk, bv, bqkv);

  // QKV projection: [8192,1024] @ [1024,3072] + bias -> bf16
  wmma_gemm<0><<<dim3(QKV_PITCH / 128, ROWS / 128), 256, 0, stream>>>(
      Xb, Wqkvb, bqkv, QKV, ROWS, QKV_PITCH, D_MODEL);

  // Flash attention, output in reference (t*H + h) channel order, fp32
  attn_kernel<<<dim3(N_TOK / 64, H_HEADS, B_SZ), 128, 0, stream>>>(QKV, AO);

  // h1 = LN(attn + tokens): f32 (residual base) + bf16 (GEMM input)
  ln_kernel<<<ROWS, 256, 0, stream>>>(AO, tokens, g1, be1, H1F, H1B);

  // ff = relu(h1 @ W1 + b1) @ W2 + b2
  wmma_gemm<1><<<dim3(DOUT / 128, ROWS / 128), 256, 0, stream>>>(
      H1B, W1b, b1, R1B, ROWS, DOUT, DOUT);
  wmma_gemm<2><<<dim3(DOUT / 128, ROWS / 128), 256, 0, stream>>>(
      R1B, W2b, b2, FF, ROWS, DOUT, DOUT);

  // out = LN(h1 + ff)
  ln_kernel<<<ROWS, 256, 0, stream>>>(H1F, FF, g2, be2, (float*)d_out, nullptr);
}
